// net_2_37228776522059
// MI455X (gfx1250) — matrix operand order repaired and checked by reading, not compiled, not run on hardware
//
#include <hip/hip_runtime.h>
#include <hip/hip_bf16.h>

typedef __attribute__((ext_vector_type(16))) __bf16 v16bf;
typedef __attribute__((ext_vector_type(8)))  float  v8f;

#define Bn    64
#define Him   256
#define Wim   256
#define IMGSZ (Him*Wim)           // 65536
#define P1    125                 // conv1 pooled H/W
#define P2    60                  // conv2 pooled H/W
#define KFC   36000               // 10*60*60
#define NFC1  1024

// ---- workspace layout (in floats) ----
#define OFF_C1   ((size_t)0)                       // 64*8*125*125 = 8,000,000
#define OFF_ACT  (OFF_C1 + (size_t)Bn*8*P1*P1)     // 64*36000     = 2,304,000
#define OFF_H1   (OFF_ACT + (size_t)Bn*KFC)        // 64*1024
#define OFF_H2   (OFF_H1 + (size_t)Bn*NFC1)        // 64*32
#define OFF_SZ   (OFF_H2 + (size_t)Bn*32)          // 64*4
#define OFF_TH   (OFF_SZ + (size_t)Bn*4)           // 64*4
#define OFF_MIN  (OFF_TH + (size_t)Bn*4)           // 1 uint

// ============================ conv1 + pool + relu ============================
// in: (64,1,256,256); w:(8,1,7,7); out pooled (64,8,125,125)
__global__ void conv1_kernel(const float* __restrict__ im,
                             const float* __restrict__ w,
                             const float* __restrict__ bias,
                             float* __restrict__ out) {
    int idx = blockIdx.x * blockDim.x + threadIdx.x;
    if (idx >= Bn * P1 * P1) return;
    int b  = idx / (P1 * P1);
    int r  = idx % (P1 * P1);
    int ph = r / P1, pw = r % P1;
    const float* img = im + (size_t)b * IMGSZ;
    int iy0 = 2 * ph, ix0 = 2 * pw;
    float p[8][8];
#pragma unroll
    for (int dy = 0; dy < 8; ++dy)
#pragma unroll
        for (int dx = 0; dx < 8; ++dx)
            p[dy][dx] = img[(iy0 + dy) * Wim + ix0 + dx];
    for (int oc = 0; oc < 8; ++oc) {
        float s00 = 0.f, s01 = 0.f, s10 = 0.f, s11 = 0.f;
        const float* wk = w + oc * 49;
#pragma unroll
        for (int ky = 0; ky < 7; ++ky)
#pragma unroll
            for (int kx = 0; kx < 7; ++kx) {
                float wv = wk[ky * 7 + kx];
                s00 = fmaf(wv, p[ky][kx],     s00);
                s01 = fmaf(wv, p[ky][kx + 1], s01);
                s10 = fmaf(wv, p[ky + 1][kx], s10);
                s11 = fmaf(wv, p[ky + 1][kx + 1], s11);
            }
        float m = fmaxf(fmaxf(s00, s01), fmaxf(s10, s11)) + bias[oc];
        out[((size_t)b * 8 + oc) * (P1 * P1) + ph * P1 + pw] = fmaxf(m, 0.f);
    }
}

// ============================ conv2 + pool + relu ============================
// in pooled (64,8,125,125); w:(10,8,5,5); out activations (64, 36000) (c*3600+y*60+x)
__global__ void conv2_kernel(const float* __restrict__ in,
                             const float* __restrict__ w,
                             const float* __restrict__ bias,
                             float* __restrict__ act) {
    int idx = blockIdx.x * blockDim.x + threadIdx.x;
    if (idx >= Bn * P2 * P2) return;
    int b  = idx / (P2 * P2);
    int r  = idx % (P2 * P2);
    int ph = r / P2, pw = r % P2;
    float acc[10][4];
#pragma unroll
    for (int oc = 0; oc < 10; ++oc)
#pragma unroll
        for (int j = 0; j < 4; ++j) acc[oc][j] = 0.f;
    int iy0 = 2 * ph, ix0 = 2 * pw;
    for (int c = 0; c < 8; ++c) {
        const float* ch = in + ((size_t)b * 8 + c) * (P1 * P1);
        float q[6][6];
#pragma unroll
        for (int dy = 0; dy < 6; ++dy)
#pragma unroll
            for (int dx = 0; dx < 6; ++dx)
                q[dy][dx] = ch[(iy0 + dy) * P1 + ix0 + dx];
        for (int oc = 0; oc < 10; ++oc) {
            const float* wk = w + ((oc * 8 + c) * 25);
#pragma unroll
            for (int ky = 0; ky < 5; ++ky)
#pragma unroll
                for (int kx = 0; kx < 5; ++kx) {
                    float wv = wk[ky * 5 + kx];
                    acc[oc][0] = fmaf(wv, q[ky][kx],         acc[oc][0]);
                    acc[oc][1] = fmaf(wv, q[ky][kx + 1],     acc[oc][1]);
                    acc[oc][2] = fmaf(wv, q[ky + 1][kx],     acc[oc][2]);
                    acc[oc][3] = fmaf(wv, q[ky + 1][kx + 1], acc[oc][3]);
                }
        }
    }
    for (int oc = 0; oc < 10; ++oc) {
        float m = fmaxf(fmaxf(acc[oc][0], acc[oc][1]), fmaxf(acc[oc][2], acc[oc][3])) + bias[oc];
        act[(size_t)b * KFC + oc * (P2 * P2) + ph * P2 + pw] = fmaxf(m, 0.f);
    }
}

// ============================ fc1 GEMM via WMMA (bf16) ============================
// A = act (64 x 36000), B = fc1_w (1024 x 36000)  (C = A * B^T), relu(+bias) -> h1 (64x1024)
__device__ inline v16bf cvt16(const float* x) {
    v16bf r;
#pragma unroll
    for (int i = 0; i < 16; ++i) r[i] = (__bf16)x[i];
    return r;
}

// Load one 32-wide K slice of A/B fragments (fp32) per the gfx1250 bf16 WMMA
// VGPR layout (A: lanes 0-15 K in {0..7,16..23}, lanes 16-31 K in {8..15,24..31};
// B: lane = N, same K split as A).
__device__ inline void load_tile(const float* __restrict__ arow,
                                 const float* __restrict__ brow,
                                 int k0, int half,
                                 float* __restrict__ ax, float* __restrict__ bx) {
    const float4* pa1 = (const float4*)(arow + k0 + 8 * half);
    const float4* pa2 = (const float4*)(arow + k0 + 16 + 8 * half);
    const float4* pb  = (const float4*)(brow + k0 + 8 * half);
    const float4* pb2 = (const float4*)(brow + k0 + 16 + 8 * half);
    float4 f;
    f = pa1[0]; ax[0]=f.x; ax[1]=f.y; ax[2]=f.z; ax[3]=f.w;
    f = pa1[1]; ax[4]=f.x; ax[5]=f.y; ax[6]=f.z; ax[7]=f.w;
    f = pa2[0]; ax[8]=f.x; ax[9]=f.y; ax[10]=f.z; ax[11]=f.w;
    f = pa2[1]; ax[12]=f.x; ax[13]=f.y; ax[14]=f.z; ax[15]=f.w;
    f = pb[0];  bx[0]=f.x; bx[1]=f.y; bx[2]=f.z; bx[3]=f.w;
    f = pb[1];  bx[4]=f.x; bx[5]=f.y; bx[6]=f.z; bx[7]=f.w;
    f = pb2[0]; bx[8]=f.x; bx[9]=f.y; bx[10]=f.z; bx[11]=f.w;
    f = pb2[1]; bx[12]=f.x; bx[13]=f.y; bx[14]=f.z; bx[15]=f.w;
}

__global__ void fc1_wmma_kernel(const float* __restrict__ act,
                                const float* __restrict__ wfc,
                                const float* __restrict__ bias,
                                float* __restrict__ h1) {
    const int lane  = threadIdx.x & 31;
    const int half  = lane >> 4;          // 0: lanes 0-15, 1: lanes 16-31
    const int l15   = lane & 15;
    const int wave  = (blockIdx.x * (blockDim.x >> 5)) + (threadIdx.x >> 5); // 0..255
    const int mt    = wave >> 6;          // 0..3   (M tile)
    const int nt    = wave & 63;          // 0..63  (N tile)
    const int m0    = mt * 16;
    const int n0    = nt * 16;

    const float* arow = act + (size_t)(m0 + l15) * KFC;
    const float* brow = wfc + (size_t)(n0 + l15) * KFC;

    v8f c = {};
    float ax0[16], bx0[16], ax1[16], bx1[16];

    // K = 36000 = 32 + 562*64 : software-pipelined ping-pong double buffer so
    // loads for tile k+32 are in flight while the WMMA for tile k executes.
    load_tile(arow, brow, 0, half, ax0, bx0);
    int k = 32;
    for (int it = 0; it < 562; ++it) {
        __builtin_prefetch(brow + k + 512, 0, 1);   // global_prefetch_b8, ~2KB ahead
        load_tile(arow, brow, k, half, ax1, bx1);
        {
            v16bf a = cvt16(ax0), b = cvt16(bx0);
            c = __builtin_amdgcn_wmma_f32_16x16x32_bf16(
                    false, a, false, b, (short)0, c, false, false);
        }
        load_tile(arow, brow, k + 32, half, ax0, bx0);
        {
            v16bf a = cvt16(ax1), b = cvt16(bx1);
            c = __builtin_amdgcn_wmma_f32_16x16x32_bf16(
                    false, a, false, b, (short)0, c, false, false);
        }
        k += 64;
    }
    {   // epilogue: consumes tile at K = 35968
        v16bf a = cvt16(ax0), b = cvt16(bx0);
        c = __builtin_amdgcn_wmma_f32_16x16x32_bf16(
                false, a, false, b, (short)0, c, false, false);
    }

    // C layout: element r, lane: M = m0 + r + 8*half, N = n0 + l15
    const int n = n0 + l15;
    const float bn = bias[n];
#pragma unroll
    for (int r = 0; r < 8; ++r) {
        int m = m0 + r + 8 * half;
        h1[(size_t)m * NFC1 + n] = fmaxf(c[r] + bn, 0.f);
    }
}

// ============================ fc2 ============================
__global__ void fc2_kernel(const float* __restrict__ h1,
                           const float* __restrict__ w,
                           const float* __restrict__ bias,
                           float* __restrict__ h2) {
    int idx = blockIdx.x * blockDim.x + threadIdx.x;
    if (idx >= Bn * 32) return;
    int b = idx >> 5, n = idx & 31;
    const float4* hv = (const float4*)(h1 + (size_t)b * NFC1);
    const float4* wv = (const float4*)(w + (size_t)n * NFC1);
    float s = 0.f;
#pragma unroll 4
    for (int i = 0; i < NFC1 / 4; ++i) {
        float4 a = hv[i], bb = wv[i];
        s = fmaf(a.x, bb.x, s); s = fmaf(a.y, bb.y, s);
        s = fmaf(a.z, bb.z, s); s = fmaf(a.w, bb.w, s);
    }
    h2[idx] = fmaxf(s + bias[n], 0.f);
}

// ============================ fc3 + theta ============================
__global__ void fc3_theta_kernel(const float* __restrict__ h2,
                                 const float* __restrict__ w3,
                                 const float* __restrict__ b3,
                                 float* __restrict__ sz,
                                 float* __restrict__ th,
                                 float* __restrict__ theta_out) {
    int b = threadIdx.x;
    if (b >= Bn) return;
    float s[4];
#pragma unroll
    for (int j = 0; j < 4; ++j) {
        float acc = 0.f;
        for (int k = 0; k < 32; ++k) acc = fmaf(h2[b * 32 + k], w3[j * 32 + k], acc);
        float v = acc + b3[j];
        s[j] = fminf(fmaxf(v, 0.f), (float)(Him - 1));
        sz[b * 4 + j] = s[j];
    }
    float t00 = (s[1] - s[0]) / (float)Him + 1e-20f;
    float t02 = (s[1] + s[0]) / (float)Him - 1.f;
    float t11 = (s[3] - s[2]) / (float)Wim + 1e-20f;
    float t12 = (s[3] + s[2]) / (float)Wim - 1.f;
    th[b * 4 + 0] = t00; th[b * 4 + 1] = t02;
    th[b * 4 + 2] = t11; th[b * 4 + 3] = t12;
    float* to = theta_out + b * 6;
    to[0] = t00; to[1] = 0.f; to[2] = t02;
    to[3] = 0.f; to[4] = t11; to[5] = t12;
}

// ============================ global min ============================
__device__ inline unsigned enc_f(float f) {
    unsigned u = __float_as_uint(f);
    return (u & 0x80000000u) ? ~u : (u | 0x80000000u);
}
__device__ inline float dec_f(unsigned e) {
    return (e & 0x80000000u) ? __uint_as_float(e & 0x7FFFFFFFu)
                             : __uint_as_float(~e);
}

__global__ void min_init_kernel(unsigned* m) { *m = 0xFFFFFFFFu; }

__global__ void min_reduce_kernel(const float* __restrict__ im, unsigned* m) {
    __shared__ unsigned sm[256];
    unsigned best = 0xFFFFFFFFu;
    size_t total = (size_t)Bn * IMGSZ;
    for (size_t i = (size_t)blockIdx.x * blockDim.x + threadIdx.x; i < total;
         i += (size_t)gridDim.x * blockDim.x) {
        unsigned e = enc_f(im[i]);
        best = e < best ? e : best;
    }
    sm[threadIdx.x] = best;
    __syncthreads();
    for (int off = 128; off > 0; off >>= 1) {
        if (threadIdx.x < off) {
            unsigned o = sm[threadIdx.x + off];
            if (o < sm[threadIdx.x]) sm[threadIdx.x] = o;
        }
        __syncthreads();
    }
    if (threadIdx.x == 0) atomicMin(m, sm[0]);
}

// ============================ sampler ============================
__global__ void sampler_kernel(const float* __restrict__ im,
                               const float* __restrict__ sz,
                               const float* __restrict__ th,
                               const unsigned* __restrict__ menc,
                               float* __restrict__ out) {
    int idx = blockIdx.x * blockDim.x + threadIdx.x;
    if (idx >= Bn * IMGSZ) return;
    int b = idx >> 16;
    int pix = idx & (IMGSZ - 1);
    int Y = pix >> 8, X = pix & 255;

    float s0 = sz[b * 4 + 0], s1 = sz[b * 4 + 1];
    float s2 = sz[b * 4 + 2], s3 = sz[b * 4 + 3];
    float t00 = th[b * 4 + 0], t02 = th[b * 4 + 1];
    float t11 = th[b * 4 + 2], t12 = th[b * 4 + 3];

    float x0 = floorf(s0), x1 = floorf(s1);
    float y0 = floorf(s2), y1 = floorf(s3);
    float w_out = x1 - x0 + 1.f, h_out = y1 - y0 + 1.f;

    float u = (2.f * ((float)X - x0) + 1.f) / w_out - 1.f;
    float v = (2.f * ((float)Y - y0) + 1.f) / h_out - 1.f;
    float gx = t00 * u + t02;
    float gy = t11 * v + t12;
    float ix = ((gx + 1.f) * (float)Wim - 1.f) * 0.5f;
    float iy = ((gy + 1.f) * (float)Him - 1.f) * 0.5f;

    const float* img = im + (size_t)b * IMGSZ;
    float xf0 = floorf(ix), yf0 = floorf(iy);
    float wx = ix - xf0, wy = iy - yf0;

    auto gather = [&](float xf, float yf) -> float {
        bool valid = (xf >= 0.f) && (xf <= (float)(Wim - 1)) &&
                     (yf >= 0.f) && (yf <= (float)(Him - 1));
        int xi = (int)fminf(fmaxf(xf, 0.f), (float)(Wim - 1));
        int yi = (int)fminf(fmaxf(yf, 0.f), (float)(Him - 1));
        return valid ? img[yi * Wim + xi] : 0.f;
    };
    float v00 = gather(xf0,       yf0);
    float v10 = gather(xf0 + 1.f, yf0);
    float v01 = gather(xf0,       yf0 + 1.f);
    float v11 = gather(xf0 + 1.f, yf0 + 1.f);
    float top = v00 * (1.f - wx) + v10 * wx;
    float bot = v01 * (1.f - wx) + v11 * wx;
    float sampled = top * (1.f - wy) + bot * wy;

    bool in_box = ((float)X >= x0) && ((float)X <= x1) &&
                  ((float)Y >= y0) && ((float)Y <= y1);
    float min_val = dec_f(*menc);
    out[idx] = in_box ? sampled : min_val;
}

// ============================ launch ============================
extern "C" void kernel_launch(void* const* d_in, const int* in_sizes, int n_in,
                              void* d_out, int out_size, void* d_ws, size_t ws_size,
                              hipStream_t stream) {
    const float* im      = (const float*)d_in[0];
    const float* conv1_w = (const float*)d_in[1];
    const float* conv1_b = (const float*)d_in[2];
    const float* conv2_w = (const float*)d_in[3];
    const float* conv2_b = (const float*)d_in[4];
    const float* fc1_w   = (const float*)d_in[5];
    const float* fc1_b   = (const float*)d_in[6];
    const float* fc2_w   = (const float*)d_in[7];
    const float* fc2_b   = (const float*)d_in[8];
    const float* fc3_w   = (const float*)d_in[9];
    const float* fc3_b   = (const float*)d_in[10];

    float* ws   = (float*)d_ws;
    float* c1   = ws + OFF_C1;
    float* act  = ws + OFF_ACT;
    float* h1   = ws + OFF_H1;
    float* h2   = ws + OFF_H2;
    float* sz   = ws + OFF_SZ;
    float* th   = ws + OFF_TH;
    unsigned* me = (unsigned*)(ws + OFF_MIN);

    float* img_out   = (float*)d_out;
    float* theta_out = (float*)d_out + (size_t)Bn * IMGSZ;

    // localization CNN
    conv1_kernel<<<(Bn * P1 * P1 + 255) / 256, 256, 0, stream>>>(im, conv1_w, conv1_b, c1);
    conv2_kernel<<<(Bn * P2 * P2 + 255) / 256, 256, 0, stream>>>(c1, conv2_w, conv2_b, act);
    fc1_wmma_kernel<<<64, 128, 0, stream>>>(act, fc1_w, fc1_b, h1);
    fc2_kernel<<<(Bn * 32 + 255) / 256, 256, 0, stream>>>(h1, fc2_w, fc2_b, h2);
    fc3_theta_kernel<<<1, 64, 0, stream>>>(h2, fc3_w, fc3_b, sz, th, theta_out);

    // global min
    min_init_kernel<<<1, 1, 0, stream>>>(me);
    min_reduce_kernel<<<2048, 256, 0, stream>>>(im, me);

    // crop / sample
    sampler_kernel<<<(Bn * IMGSZ + 255) / 256, 256, 0, stream>>>(im, sz, th, me, img_out);
}
